// ImageCaptioningModel_65189013618975
// MI455X (gfx1250) — matrix operand order repaired and checked by reading, not compiled, not run on hardware
//
#include <hip/hip_runtime.h>
#include <hip/hip_bf16.h>

// ---------------- problem constants ----------------
#define BSZ   32
#define PPIX  196          // 14*14
#define ENC   2048
#define EMB   512
#define DEC   512
#define ATT   512
#define VOC   10000
#define LCAP  52
#define TDEC  51           // L-1
#define KG    3072         // EMB + ENC + DEC  (fused [emb|awe|h] @ [W_ih|W_hh]^T)
#define NCOMB 2560         // [W_dec ; W_beta] rows

// ---------------- d_out flat layout (floats, reference return order) ----
#define PRED_OFF  ((size_t)0)                              // 32*51*10000
#define TOKS_OFF  ((size_t)(BSZ*TDEC*VOC))                 // 32*52
#define DLEN_OFF  (TOKS_OFF + (size_t)BSZ*LCAP)            // 32
#define ALPH_OFF  (DLEN_OFF + (size_t)BSZ)                 // 32*51*196
#define SORT_OFF  (ALPH_OFF + (size_t)BSZ*TDEC*PPIX)       // 32

// ---------------- workspace byte offsets ----------------
#define OFF_SORT  ((size_t)0)                    // int[32]
#define OFF_DLEN  ((size_t)256)                  // int[32]
#define OFF_TOKS  ((size_t)512)                  // int[32*52]
#define OFF_H     ((size_t)8192)                 // f32[32*512]
#define OFF_C     (OFF_H    + (size_t)BSZ*DEC*4)
#define OFF_G     (OFF_C    + (size_t)BSZ*DEC*4)          // f32[32*2048]
#define OFF_XCAT  (OFF_G    + (size_t)BSZ*4*DEC*4)        // bf16[32*3072]
#define OFF_HBF   (OFF_XCAT + (size_t)BSZ*KG*2)           // bf16[32*512] (h_new, for FC)
#define OFF_HSB   (OFF_HBF  + (size_t)BSZ*DEC*2)          // bf16[32*512] (frozen h state)
#define OFF_EMBB  (OFF_HSB  + (size_t)BSZ*DEC*2)          // bf16[32*52*512]
#define OFF_WG    (OFF_EMBB + (size_t)BSZ*LCAP*EMB*2)     // bf16[2048*3072]
#define OFF_WFC   (OFF_WG   + (size_t)(4*DEC)*KG*2)       // bf16[10000*512]
#define OFF_WCOMB (OFF_WFC  + (size_t)VOC*DEC*2)          // bf16[2560*512]
#define OFF_GD    (OFF_WCOMB+ (size_t)NCOMB*DEC*2)        // f32[32*2560] ([dec_att|gate_pre])
#define OFF_ENC   (OFF_GD   + (size_t)BSZ*NCOMB*4)        // f32[6272*512]
// total ~41 MB

// ---------------- optional gfx1250 async-to-LDS path ----------------
#if defined(__has_builtin)
#if __has_builtin(__builtin_amdgcn_global_load_async_to_lds_b128) && \
    __has_builtin(__builtin_amdgcn_s_wait_asynccnt)
#define HAVE_ASYNC_LDS 1
#endif
#endif

#if HAVE_ASYNC_LDS
// exact parameter type from hipcc diagnostic: int __vector(4) in AS1 / AS3
typedef int v4i_gcc __attribute__((vector_size(16)));
typedef __attribute__((address_space(1))) v4i_gcc* async_glb_p;
typedef __attribute__((address_space(3))) v4i_gcc* async_lds_p;
#endif

// ---------------- WMMA types / helpers ----------------
typedef __attribute__((ext_vector_type(16))) __bf16         bfrag;
typedef __attribute__((ext_vector_type(8)))  float          v8f;
typedef __attribute__((ext_vector_type(16))) unsigned short v16us;

struct U8x32 { uint4 a, b; };   // 32 bytes, bit-pattern for a 16x bf16 fragment

static __device__ __forceinline__ unsigned short f2bfbits(float f) {
    unsigned int x = __float_as_uint(f);
    x += 0x7fffu + ((x >> 16) & 1u);     // round-to-nearest-even
    return (unsigned short)(x >> 16);
}
static __device__ __forceinline__ float sigmf(float x) { return 1.f / (1.f + expf(-x)); }

// A-fragment (16x32 bf16, ISA 7.12.2): lane<16 holds row m0+lane, K in
// {k0..k0+7, k0+16..k0+23}; lane>=16 same rows shifted by +8 in K.
// Works for global or LDS (generic) pointers.
static __device__ __forceinline__ bfrag load_a_bf16(const unsigned short* A, int lda,
                                                    int m0, int k0, int lane) {
    int hi = lane >> 4;
    const unsigned short* p = A + (size_t)(m0 + (lane & 15)) * lda + k0 + hi * 8;
    U8x32 r{ *(const uint4*)p, *(const uint4*)(p + 16) };
    return __builtin_bit_cast(bfrag, r);
}
// B-fragment (32x16 bf16) for D = A * W^T with W row-major (N x K).
static __device__ __forceinline__ bfrag load_b_bf16(const unsigned short* W, int ldw,
                                                    int n0, int k0, int lane) {
    int hi = lane >> 4;
    const unsigned short* p = W + (size_t)(n0 + (lane & 15)) * ldw + k0 + hi * 8;
    U8x32 r{ *(const uint4*)p, *(const uint4*)(p + 16) };
    return __builtin_bit_cast(bfrag, r);
}
// fp32-source variants (convert on the fly; one-time enc_att GEMM only)
static __device__ __forceinline__ bfrag load_a_f32row(const float* row, int k0, int lane) {
    int hi = lane >> 4;
    const float* p = row + k0 + hi * 8;
    v16us u;
#pragma unroll
    for (int i = 0; i < 8; ++i) { u[i] = f2bfbits(p[i]); u[i + 8] = f2bfbits(p[i + 16]); }
    return __builtin_bit_cast(bfrag, u);
}
static __device__ __forceinline__ bfrag load_b_f32(const float* W, int ldw,
                                                   int n0, int k0, int lane) {
    int hi = lane >> 4;
    const float* p = W + (size_t)(n0 + (lane & 15)) * ldw + k0 + hi * 8;
    v16us u;
#pragma unroll
    for (int i = 0; i < 8; ++i) { u[i] = f2bfbits(p[i]); u[i + 8] = f2bfbits(p[i + 16]); }
    return __builtin_bit_cast(bfrag, u);
}
static __device__ __forceinline__ v8f wmma_bf16(bfrag a, bfrag b, v8f c) {
    return __builtin_amdgcn_wmma_f32_16x16x32_bf16(false, a, false, b, (short)0, c, false, false);
}

// ================= setup kernels =================

// stable descending sort of caption_lengths (B=32, single thread is fine)
__global__ void k_sort(const int* __restrict__ lengths, const int* __restrict__ toks,
                       float* __restrict__ out, int* __restrict__ wsSort,
                       int* __restrict__ wsDlen, int* __restrict__ wsToks) {
    if (threadIdx.x != 0 || blockIdx.x != 0) return;
    bool taken[BSZ];
    for (int i = 0; i < BSZ; ++i) taken[i] = false;
    for (int i = 0; i < BSZ; ++i) {
        int best = -1, bl = -0x7fffffff;
        for (int j = 0; j < BSZ; ++j)
            if (!taken[j] && lengths[j] > bl) { bl = lengths[j]; best = j; }  // '>' keeps stability
        taken[best] = true;
        wsSort[i] = best;
        out[SORT_OFF + i] = (float)best;
        int dl = lengths[best] - 1;
        wsDlen[i] = dl;
        out[DLEN_OFF + i] = (float)dl;
        for (int l = 0; l < LCAP; ++l) {
            int tk = toks[best * LCAP + l];
            wsToks[i * LCAP + l] = tk;
            out[TOKS_OFF + (size_t)i * LCAP + l] = (float)tk;
        }
    }
}

// h0/c0 from mean image feature (fp32, small); also emit bf16 h-state
__global__ void k_h0c0(const float* __restrict__ img, const float* __restrict__ Wh0,
                       const float* __restrict__ bh0, const float* __restrict__ Wc0,
                       const float* __restrict__ bc0, const int* __restrict__ wsSort,
                       float* __restrict__ h, float* __restrict__ c,
                       unsigned short* __restrict__ hsb) {
    __shared__ float mean[ENC];
    int b = blockIdx.x, src = wsSort[b];
    const float* base = img + (size_t)src * PPIX * ENC;
    for (int e = threadIdx.x; e < ENC; e += blockDim.x) {
        float s = 0.f;
        for (int p = 0; p < PPIX; ++p) s += base[(size_t)p * ENC + e];
        mean[e] = s * (1.f / PPIX);
    }
    __syncthreads();
    for (int j = threadIdx.x; j < DEC; j += blockDim.x) {
        float sh = bh0[j], sc = bc0[j];
        const float* wh = Wh0 + (size_t)j * ENC;
        const float* wc = Wc0 + (size_t)j * ENC;
        for (int e = 0; e < ENC; ++e) { sh += mean[e] * wh[e]; sc += mean[e] * wc[e]; }
        h[b * DEC + j] = sh;
        c[b * DEC + j] = sc;
        hsb[b * DEC + j] = f2bfbits(sh);
    }
}

// pack [W_ih | W_hh] into one bf16 (2048 x 3072) weight, reused 51x from L2
__global__ void k_cvt_wg(const float* __restrict__ Wih, const float* __restrict__ Whh,
                         unsigned short* __restrict__ Wg) {
    size_t n = (size_t)(4 * DEC) * KG;
    for (size_t i = (size_t)blockIdx.x * blockDim.x + threadIdx.x; i < n;
         i += (size_t)gridDim.x * blockDim.x) {
        size_t r = i / KG, cidx = i % KG;
        float v = (cidx < (EMB + ENC)) ? Wih[r * (EMB + ENC) + cidx]
                                       : Whh[r * DEC + (cidx - (EMB + ENC))];
        Wg[i] = f2bfbits(v);
    }
}
__global__ void k_cvt_wfc(const float* __restrict__ Wfc, unsigned short* __restrict__ W) {
    size_t n = (size_t)VOC * DEC;
    for (size_t i = (size_t)blockIdx.x * blockDim.x + threadIdx.x; i < n;
         i += (size_t)gridDim.x * blockDim.x)
        W[i] = f2bfbits(Wfc[i]);
}
// pack [W_dec ; W_beta] (2560 x 512) bf16 for the per-step gate/dec GEMM
__global__ void k_cvt_wcomb(const float* __restrict__ Wdec, const float* __restrict__ Wbeta,
                            unsigned short* __restrict__ W) {
    size_t n = (size_t)NCOMB * DEC;
    for (size_t i = (size_t)blockIdx.x * blockDim.x + threadIdx.x; i < n;
         i += (size_t)gridDim.x * blockDim.x) {
        size_t r = i / DEC, cidx = i % DEC;
        float v = (r < DEC) ? Wdec[r * DEC + cidx] : Wbeta[(r - DEC) * DEC + cidx];
        W[i] = f2bfbits(v);
    }
}
// gather embeddings for sorted tokens, bf16
__global__ void k_embs(const float* __restrict__ table, const int* __restrict__ toks,
                       unsigned short* __restrict__ embb) {
    size_t n = (size_t)BSZ * LCAP * EMB;
    for (size_t i = (size_t)blockIdx.x * blockDim.x + threadIdx.x; i < n;
         i += (size_t)gridDim.x * blockDim.x) {
        size_t bl = i / EMB, e = i % EMB;
        embb[i] = f2bfbits(table[(size_t)toks[bl] * EMB + e]);
    }
}

// one-time encoder projection: enc_att(6272x512) = img_sorted @ W_enc^T.
// 2 N-tiles per wave: the fp32->bf16 A conversion is amortized over 2 WMMAs.
__global__ void k_encatt(const float* __restrict__ img, const float* __restrict__ Wenc,
                         const float* __restrict__ benc, const int* __restrict__ wsSort,
                         float* __restrict__ encatt) {
    int lane = threadIdx.x & 31, hi = lane >> 4, lm = lane & 15;
    int n0 = blockIdx.x * 32, m0 = blockIdx.y * 16;
    int r = m0 + lm, bb = r / PPIX, pp = r % PPIX;
    const float* arow = img + ((size_t)wsSort[bb] * PPIX + pp) * ENC;
    v8f acc0 = {}, acc1 = {};
    for (int k0 = 0; k0 < ENC; k0 += 32) {
        bfrag a  = load_a_f32row(arow, k0, lane);
        bfrag b0 = load_b_f32(Wenc, ENC, n0,      k0, lane);
        bfrag b1 = load_b_f32(Wenc, ENC, n0 + 16, k0, lane);
        acc0 = wmma_bf16(a, b0, acc0);
        acc1 = wmma_bf16(a, b1, acc1);
    }
#pragma unroll
    for (int rr = 0; rr < 8; ++rr) {
        int m = rr + hi * 8;
        int na = n0 + lm, nb = n0 + 16 + lm;
        encatt[(size_t)(m0 + m) * ATT + na] = acc0[rr] + benc[na];
        encatt[(size_t)(m0 + m) * ATT + nb] = acc1[rr] + benc[nb];
    }
}

// ================= per-step kernels =================

// [dec_att | gate_pre](32x2560) = h @ [W_dec;W_beta]^T + bias   (WMMA)
// 4 waves/block, each wave owns 1 N-tile and both M-tiles (2 indep. acc chains)
__global__ void k_gatedec(const unsigned short* __restrict__ hsb,
                          const unsigned short* __restrict__ Wcomb,
                          const float* __restrict__ bdec, const float* __restrict__ bbeta,
                          float* __restrict__ gd) {
    int lane = threadIdx.x & 31, hi = lane >> 4, lm = lane & 15;
    int nt = blockIdx.x * 4 + (threadIdx.x >> 5);
    int n0 = nt * 16;
    v8f acc0 = {}, acc1 = {};
    for (int k0 = 0; k0 < DEC; k0 += 32) {
        bfrag b  = load_b_bf16(Wcomb, DEC, n0, k0, lane);
        bfrag a0 = load_a_bf16(hsb, DEC, 0,  k0, lane);
        bfrag a1 = load_a_bf16(hsb, DEC, 16, k0, lane);
        acc0 = wmma_bf16(a0, b, acc0);
        acc1 = wmma_bf16(a1, b, acc1);
    }
    int n = n0 + lm;
    float bias = (n < DEC) ? bdec[n] : bbeta[n - DEC];
#pragma unroll
    for (int rr = 0; rr < 8; ++rr) {
        int m = rr + hi * 8;
        gd[(size_t)m        * NCOMB + n] = acc0[rr] + bias;
        gd[(size_t)(m + 16) * NCOMB + n] = acc1[rr] + bias;
    }
}

// attention core: score -> softmax -> awe*gate; assemble xcat=[emb|gate*awe|h]
__global__ void k_attn(int t, const float* __restrict__ img, const float* __restrict__ wfull,
                       const float* __restrict__ bfull, const int* __restrict__ wsSort,
                       const int* __restrict__ wsDlen, const float* __restrict__ encatt,
                       const float* __restrict__ gd, const unsigned short* __restrict__ hsb,
                       const unsigned short* __restrict__ embb,
                       unsigned short* __restrict__ xcat, float* __restrict__ out) {
    __shared__ float sh_da[ATT];
    __shared__ float sh_sc[PPIX];
    __shared__ float s_inv;
    int b = blockIdx.x, tid = threadIdx.x;
    int src = wsSort[b];
    bool act = (t < wsDlen[b]);
    for (int j = tid; j < ATT; j += blockDim.x) sh_da[j] = gd[(size_t)b * NCOMB + j];
    __syncthreads();
    for (int p = tid; p < PPIX; p += blockDim.x) {
        const float* ea = encatt + ((size_t)b * PPIX + p) * ATT;
        float s = 0.f;
        for (int a = 0; a < ATT; ++a) s += fmaxf(ea[a] + sh_da[a], 0.f) * wfull[a];
        sh_sc[p] = s + bfull[0];
    }
    __syncthreads();
    if (tid == 0) {
        float mx = -1e30f;
        for (int p = 0; p < PPIX; ++p) mx = fmaxf(mx, sh_sc[p]);
        float sm = 0.f;
        for (int p = 0; p < PPIX; ++p) { float e = expf(sh_sc[p] - mx); sh_sc[p] = e; sm += e; }
        s_inv = 1.f / sm;
    }
    __syncthreads();
    for (int p = tid; p < PPIX; p += blockDim.x) {
        float al = sh_sc[p] * s_inv;
        sh_sc[p] = al;
        out[ALPH_OFF + ((size_t)b * TDEC + t) * PPIX + p] = act ? al : 0.f;
    }
    __syncthreads();
    const float* ibase = img + (size_t)src * PPIX * ENC;
    for (int e = tid; e < ENC; e += blockDim.x) {
        float aw = 0.f;
        for (int p = 0; p < PPIX; ++p) aw += sh_sc[p] * ibase[(size_t)p * ENC + e];
        float gate = sigmf(gd[(size_t)b * NCOMB + DEC + e]);
        xcat[(size_t)b * KG + EMB + e] = f2bfbits(gate * aw);
    }
    for (int j = tid; j < EMB; j += blockDim.x) {
        xcat[(size_t)b * KG + j]             = embb[((size_t)b * LCAP + t) * EMB + j];
        xcat[(size_t)b * KG + EMB + ENC + j] = hsb[b * DEC + j];
    }
}

// fused LSTM gate GEMM: g(32x2048) = xcat(32x3072) @ Wg^T + b_ih + b_hh
// 4 waves/block; each wave: 1 N-tile, 2 M-tile accumulator chains
__global__ void k_lstm_gemm(const unsigned short* __restrict__ xcat,
                            const unsigned short* __restrict__ Wg,
                            const float* __restrict__ bih, const float* __restrict__ bhh,
                            float* __restrict__ g) {
    int lane = threadIdx.x & 31, hi = lane >> 4, lm = lane & 15;
    int nt = blockIdx.x * 4 + (threadIdx.x >> 5);
    int n0 = nt * 16;
    v8f acc0 = {}, acc1 = {};
    for (int k0 = 0; k0 < KG; k0 += 32) {
        bfrag b  = load_b_bf16(Wg, KG, n0, k0, lane);
        bfrag a0 = load_a_bf16(xcat, KG, 0,  k0, lane);
        bfrag a1 = load_a_bf16(xcat, KG, 16, k0, lane);
        acc0 = wmma_bf16(a0, b, acc0);
        acc1 = wmma_bf16(a1, b, acc1);
    }
    int n = n0 + lm;
    float bias = bih[n] + bhh[n];
#pragma unroll
    for (int rr = 0; rr < 8; ++rr) {
        int m = rr + hi * 8;
        g[(size_t)m        * (4 * DEC) + n] = acc0[rr] + bias;
        g[(size_t)(m + 16) * (4 * DEC) + n] = acc1[rr] + bias;
    }
}

// LSTM pointwise with freeze-masking; emits bf16 h_new (FC) and bf16 h-state
__global__ void k_point(int t, const float* __restrict__ g, const int* __restrict__ wsDlen,
                        float* __restrict__ h, float* __restrict__ c,
                        unsigned short* __restrict__ hbf, unsigned short* __restrict__ hsb) {
    int i = blockIdx.x * blockDim.x + threadIdx.x;
    if (i >= BSZ * DEC) return;
    int b = i / DEC, jj = i % DEC;
    const float* gr = g + (size_t)b * (4 * DEC);
    float gi = gr[jj], gf = gr[DEC + jj], gg = gr[2 * DEC + jj], go = gr[3 * DEC + jj];
    float hold = h[i];
    float cn = sigmf(gf) * c[i] + sigmf(gi) * tanhf(gg);
    float hn = sigmf(go) * tanhf(cn);
    bool act = (t < wsDlen[b]);
    if (act) { h[i] = hn; c[i] = cn; }
    hbf[i] = f2bfbits(hn);                    // preds use h_new; masked in k_fc
    hsb[i] = f2bfbits(act ? hn : hold);       // frozen state for next step
}

// FC head: preds(32x10000) = h_new @ W_fc^T + b_fc, masked by active.
// A (32x512 bf16 = 32KB) staged once per 4-wave block into LDS
// (gfx1250 async-to-LDS when available), shared by 4 N-tiles.
__global__ void k_fc(int t, const unsigned short* __restrict__ hbf,
                     const unsigned short* __restrict__ Wfc, const float* __restrict__ bfc,
                     const int* __restrict__ wsDlen, float* __restrict__ out) {
    __shared__ __align__(32) unsigned short shA[BSZ * DEC];
    int tid = threadIdx.x;
#if HAVE_ASYNC_LDS
    for (int i = tid; i < (BSZ * DEC) / 8; i += blockDim.x)
        __builtin_amdgcn_global_load_async_to_lds_b128(
            (async_glb_p)(hbf + (size_t)i * 8), (async_lds_p)(shA + (size_t)i * 8), 0, 0);
    __builtin_amdgcn_s_wait_asynccnt(0);
#else
    for (int i = tid; i < (BSZ * DEC) / 8; i += blockDim.x)
        *(uint4*)(shA + (size_t)i * 8) = *(const uint4*)(hbf + (size_t)i * 8);
#endif
    __syncthreads();

    int lane = tid & 31, hi = lane >> 4, lm = lane & 15;
    int nt = blockIdx.x * 4 + (tid >> 5);
    if (nt >= VOC / 16) return;               // wave-uniform guard (EXEC stays all-1)
    int n0 = nt * 16;
    v8f acc0 = {}, acc1 = {};
    for (int k0 = 0; k0 < DEC; k0 += 32) {
        bfrag b  = load_b_bf16(Wfc, DEC, n0, k0, lane);
        bfrag a0 = load_a_bf16(shA, DEC, 0,  k0, lane);   // ds_load_b128 fragments
        bfrag a1 = load_a_bf16(shA, DEC, 16, k0, lane);
        acc0 = wmma_bf16(a0, b, acc0);
        acc1 = wmma_bf16(a1, b, acc1);
    }
    int n = n0 + lm;
    float bias = bfc[n];
#pragma unroll
    for (int rr = 0; rr < 8; ++rr) {
        int m = rr + hi * 8;
        bool a0act = (t < wsDlen[m]);
        bool a1act = (t < wsDlen[m + 16]);
        out[PRED_OFF + ((size_t)m        * TDEC + t) * VOC + n] = a0act ? (acc0[rr] + bias) : 0.f;
        out[PRED_OFF + ((size_t)(m + 16) * TDEC + t) * VOC + n] = a1act ? (acc1[rr] + bias) : 0.f;
    }
}

// ================= host driver =================
extern "C" void kernel_launch(void* const* d_in, const int* in_sizes, int n_in,
                              void* d_out, int out_size, void* d_ws, size_t ws_size,
                              hipStream_t stream) {
    const float* img    = (const float*)d_in[0];
    const int*   toks   = (const int*)d_in[1];
    const int*   lens   = (const int*)d_in[2];
    const float* Wenc   = (const float*)d_in[3];
    const float* benc   = (const float*)d_in[4];
    const float* Wdec   = (const float*)d_in[5];
    const float* bdec   = (const float*)d_in[6];
    const float* wfull  = (const float*)d_in[7];
    const float* bfull  = (const float*)d_in[8];
    const float* table  = (const float*)d_in[9];
    const float* Wih    = (const float*)d_in[10];
    const float* bih    = (const float*)d_in[11];
    const float* Whh    = (const float*)d_in[12];
    const float* bhh    = (const float*)d_in[13];
    const float* Wh0    = (const float*)d_in[14];
    const float* bh0    = (const float*)d_in[15];
    const float* Wc0    = (const float*)d_in[16];
    const float* bc0    = (const float*)d_in[17];
    const float* Wbeta  = (const float*)d_in[18];
    const float* bbeta  = (const float*)d_in[19];
    const float* Wfc    = (const float*)d_in[20];
    const float* bfc    = (const float*)d_in[21];
    float* out = (float*)d_out;

    char* ws = (char*)d_ws;
    int*            wsSort  = (int*)(ws + OFF_SORT);
    int*            wsDlen  = (int*)(ws + OFF_DLEN);
    int*            wsToks  = (int*)(ws + OFF_TOKS);
    float*          wsH     = (float*)(ws + OFF_H);
    float*          wsC     = (float*)(ws + OFF_C);
    float*          wsG     = (float*)(ws + OFF_G);
    unsigned short* wsXcat  = (unsigned short*)(ws + OFF_XCAT);
    unsigned short* wsHbf   = (unsigned short*)(ws + OFF_HBF);
    unsigned short* wsHsb   = (unsigned short*)(ws + OFF_HSB);
    unsigned short* wsEmbB  = (unsigned short*)(ws + OFF_EMBB);
    unsigned short* wsWg    = (unsigned short*)(ws + OFF_WG);
    unsigned short* wsWfc   = (unsigned short*)(ws + OFF_WFC);
    unsigned short* wsWcomb = (unsigned short*)(ws + OFF_WCOMB);
    float*          wsGd    = (float*)(ws + OFF_GD);
    float*          wsEnc   = (float*)(ws + OFF_ENC);

    // setup
    k_sort<<<1, 32, 0, stream>>>(lens, toks, out, wsSort, wsDlen, wsToks);
    k_h0c0<<<BSZ, 256, 0, stream>>>(img, Wh0, bh0, Wc0, bc0, wsSort, wsH, wsC, wsHsb);
    k_cvt_wg<<<4096, 256, 0, stream>>>(Wih, Whh, wsWg);
    k_cvt_wfc<<<4096, 256, 0, stream>>>(Wfc, wsWfc);
    k_cvt_wcomb<<<1024, 256, 0, stream>>>(Wdec, Wbeta, wsWcomb);
    k_embs<<<3328, 256, 0, stream>>>(table, wsToks, wsEmbB);
    k_encatt<<<dim3(ATT / 32, (BSZ * PPIX) / 16), 32, 0, stream>>>(img, Wenc, benc, wsSort, wsEnc);

    // sequential decode loop (graph-captured as 5*51 launches)
    for (int t = 0; t < TDEC; ++t) {
        k_gatedec<<<NCOMB / 16 / 4, 128, 0, stream>>>(wsHsb, wsWcomb, bdec, bbeta, wsGd);
        k_attn<<<BSZ, 256, 0, stream>>>(t, img, wfull, bfull, wsSort, wsDlen,
                                        wsEnc, wsGd, wsHsb, wsEmbB, wsXcat, out);
        k_lstm_gemm<<<(4 * DEC) / 16 / 4, 128, 0, stream>>>(wsXcat, wsWg, bih, bhh, wsG);
        k_point<<<(BSZ * DEC + 255) / 256, 256, 0, stream>>>(t, wsG, wsDlen, wsH, wsC,
                                                             wsHbf, wsHsb);
        k_fc<<<(VOC / 16 + 3) / 4, 128, 0, stream>>>(t, wsHbf, wsWfc, bfc, wsDlen, out);
    }
}
